// PolNetLSTM_274877906994
// MI455X (gfx1250) — compile-verified, matrix operand order checked by reading, not run on hardware
//
#include <hip/hip_runtime.h>
#include <hip/hip_bf16.h>

// ---------------------------------------------------------------------------
// PolNetLSTM for MI455X (gfx1250), wave32 + WMMA bf16 + async LDS copies.
//   Phase A: x_proj = relu(xs@W_in^T+b_in)      [WMMA, bf16 in / bf16 out]
//            gates_x = x_proj@W_ih^T+b_ih       [WMMA, bf16 in / f32 out]
//   Phase B: persistent 16-WG scan kernel, one grid barrier per timestep,
//            W_hh slice LDS-resident, h@W_hh^T via WMMA, i/f/g/o co-resident.
//   Phase C: means = tanh(hiddens@W_mean^T+b_mean)  [WMMA]
// ---------------------------------------------------------------------------

typedef __bf16 bf16;
typedef __attribute__((ext_vector_type(16))) __bf16 bf16x16;
typedef __attribute__((ext_vector_type(8)))  __bf16 bf16x8;
typedef __attribute__((ext_vector_type(8)))  float  f32x8;
typedef int v4i __attribute__((vector_size(16)));   // matches async-LDS builtin

constexpr int T_   = 512;
constexpr int B_   = 64;
constexpr int OBS_ = 128;
constexpr int H_   = 1024;
constexpr int C_   = 512;
constexpr int A_   = 32;
constexpr int TB_  = T_ * B_;          // 32768
constexpr int G4_  = 4 * C_;           // 2048
constexpr long TBA_ = (long)TB_ * A_;  // 1048576
constexpr int NWG_ = 16;               // scan workgroups

// ---- workspace layout (bytes) ---------------------------------------------
constexpr size_t OFF_XSB  = 0;                                      // xs bf16 [TB,OBS]
constexpr size_t OFF_WINB = OFF_XSB  + (size_t)TB_ * OBS_ * 2;      // W_in bf16 [H,OBS]
constexpr size_t OFF_WIHB = OFF_WINB + (size_t)H_  * OBS_ * 2;      // W_ih bf16 [4C,H]
constexpr size_t OFF_WHHB = OFF_WIHB + (size_t)G4_ * H_   * 2;      // W_hh bf16 [4C,C]
constexpr size_t OFF_WMB  = OFF_WHHB + (size_t)G4_ * C_   * 2;      // W_mean bf16 [A,C]
constexpr size_t OFF_XP   = OFF_WMB  + (size_t)A_  * C_   * 2;      // x_proj bf16 [TB,H]
constexpr size_t OFF_GX   = OFF_XP   + (size_t)TB_ * H_   * 2;      // gates_x f32 [TB,4C]
constexpr size_t OFF_HID  = OFF_GX   + (size_t)TB_ * G4_  * 4;      // hiddens bf16 [TB,C]
constexpr size_t OFF_HB   = OFF_HID  + (size_t)TB_ * C_   * 2;      // h double buf bf16 [2,B,C]
constexpr size_t OFF_BAR  = OFF_HB   + (size_t)2 * B_ * C_ * 2;     // barrier int

// ---- CDNA5 feature probes --------------------------------------------------
#if defined(__has_builtin)
#if __has_builtin(__builtin_amdgcn_global_load_async_to_lds_b128) && \
    __has_builtin(__builtin_amdgcn_s_wait_asynccnt)
#define HAS_ASYNC_LDS 1
#endif
#endif
#ifndef HAS_ASYNC_LDS
#define HAS_ASYNC_LDS 0
#endif

#define AS1 __attribute__((address_space(1)))
#define AS3 __attribute__((address_space(3)))

__device__ __forceinline__ void async_copy16(const bf16* g, bf16* l) {
#if HAS_ASYNC_LDS
  __builtin_amdgcn_global_load_async_to_lds_b128((AS1 v4i*)g, (AS3 v4i*)l, 0, 0);
#else
  *(bf16x8*)l = *(const bf16x8*)g;
#endif
}

__device__ __forceinline__ void async_wait_all() {
#if HAS_ASYNC_LDS
  __builtin_amdgcn_s_wait_asynccnt(0);
#endif
}

// ---- fast transcendentals (hardware v_exp/v_rcp/v_tanh, branch-free) -------
#if __has_builtin(__builtin_amdgcn_exp2f) && __has_builtin(__builtin_amdgcn_rcpf)
__device__ __forceinline__ float fsigmoid(float x) {
  return __builtin_amdgcn_rcpf(1.f + __builtin_amdgcn_exp2f(-1.44269504f * x));
}
#else
__device__ __forceinline__ float fsigmoid(float x) { return 1.f / (1.f + __expf(-x)); }
#endif

#if __has_builtin(__builtin_amdgcn_tanhf)
__device__ __forceinline__ float ftanh(float x) { return __builtin_amdgcn_tanhf(x); }
#elif __has_builtin(__builtin_amdgcn_exp2f) && __has_builtin(__builtin_amdgcn_rcpf)
__device__ __forceinline__ float ftanh(float x) {
  float e = __builtin_amdgcn_exp2f(2.88539008f * x);   // e^(2x)
  return 1.f - 2.f * __builtin_amdgcn_rcpf(e + 1.f);
}
#else
__device__ __forceinline__ float ftanh(float x) { return tanhf(x); }
#endif

// ---------------------------------------------------------------------------
__device__ __forceinline__ f32x8 wmma_bf16(bf16x16 a, bf16x16 b, f32x8 c) {
  return __builtin_amdgcn_wmma_f32_16x16x32_bf16(false, a, false, b,
                                                 (short)0, c, false, false);
}

// Load a 16x32 bf16 fragment (A or B^T) from row-major memory (global or LDS).
// Per ISA layout: lanes 0-15 -> rows 0-15 with K=[0..8)+[16..24),
//                 lanes 16-31 -> rows 0-15 with K=[8..16)+[24..32).
__device__ __forceinline__ bf16x16 load_frag(const bf16* __restrict__ base, int ld) {
  const int lane = threadIdx.x & 31;
  const bf16* p = base + (size_t)(lane & 15) * ld + ((lane >> 4) << 3);
  bf16x8 lo = *(const bf16x8*)(p);
  bf16x8 hi = *(const bf16x8*)(p + 16);
  bf16x16 f;
#pragma unroll
  for (int i = 0; i < 8; ++i) { f[i] = lo[i]; f[i + 8] = hi[i]; }
  return f;
}

// Device-scope grid barrier (monotonic counter; re-zeroed each launch).
__device__ __forceinline__ void grid_sync(int* bar, int target) {
  __threadfence();
  __syncthreads();
  if (threadIdx.x == 0) {
    __hip_atomic_fetch_add(bar, 1, __ATOMIC_ACQ_REL, __HIP_MEMORY_SCOPE_AGENT);
    while (__hip_atomic_load(bar, __ATOMIC_ACQUIRE, __HIP_MEMORY_SCOPE_AGENT) < target) {
      __builtin_amdgcn_s_sleep(2);
    }
  }
  __syncthreads();
}

// ---------------------------------------------------------------------------
__global__ void init_bar(int* bar) { if (threadIdx.x == 0) *bar = 0; }

__global__ void cvt_f32_bf16(const float* __restrict__ src, bf16* __restrict__ dst, long n) {
  long i = (long)blockIdx.x * blockDim.x + threadIdx.x;
  const long stride = (long)gridDim.x * blockDim.x;
  for (; i < n; i += stride) dst[i] = (bf16)src[i];
}

__global__ void fill_logstd(const float* __restrict__ ls, float* __restrict__ out, long n) {
  long i = (long)blockIdx.x * blockDim.x + threadIdx.x;
  const long stride = (long)gridDim.x * blockDim.x;
  for (; i < n; i += stride) out[i] = ls[i & (A_ - 1)];
}

// ---------------------------------------------------------------------------
// D[M,N] = act(A[M,K] @ B[N,K]^T + bias[N]); WG = 8 waves, 128M x 128N tile.
// B chunk (128x32 bf16, 8KB) double-buffered in LDS via async-to-LDS copies
// issued one K-step ahead of the WMMA consumption.
template <bool RELU, bool OUTBF>
__global__ void __launch_bounds__(256)
gemm_bias_act(const bf16* __restrict__ Am, const bf16* __restrict__ Bm,
              const float* __restrict__ bias, float* __restrict__ outF,
              bf16* __restrict__ outB, int M, int N, int K) {
  __shared__ bf16 lB[2][128 * 32];
  const int tid  = threadIdx.x;
  const int wave = tid >> 5, lane = tid & 31;
  const int row0 = blockIdx.y * 128 + wave * 16;
  const int col0 = blockIdx.x * 128;

  f32x8 acc[8];
#pragma unroll
  for (int nt = 0; nt < 8; ++nt)
#pragma unroll
    for (int i = 0; i < 8; ++i) acc[nt][i] = 0.f;

  auto stage = [&](int buf, int kc) {
#pragma unroll
    for (int it = 0; it < 2; ++it) {
      const int idx = tid + it * 256;          // 512 x 16B = 8KB
      const int n = idx >> 2, part = idx & 3;
      async_copy16(&Bm[(size_t)(col0 + n) * K + kc + part * 8],
                   &lB[buf][n * 32 + part * 8]);
    }
  };

  stage(0, 0);
  int buf = 0;
  for (int kc = 0; kc < K; kc += 32) {
    async_wait_all();          // my async fills of lB[buf] have landed
    __syncthreads();           // everyone's fills landed; prev reads done
    if (kc + 32 < K) {
      stage(buf ^ 1, kc + 32);
      __builtin_prefetch(Am + (size_t)(row0 + (lane & 15)) * K + kc + 32, 0, 3);
    }
    const bf16x16 afrag = load_frag(Am + (size_t)row0 * K + kc, K);
#pragma unroll
    for (int nt = 0; nt < 8; ++nt) {
      const bf16x16 bfrag = load_frag(&lB[buf][nt * 16 * 32], 32);
      acc[nt] = wmma_bf16(afrag, bfrag, acc[nt]);
    }
    buf ^= 1;
  }

#pragma unroll
  for (int nt = 0; nt < 8; ++nt) {
    const int col = col0 + nt * 16 + (lane & 15);
    const float bv = bias[col];
#pragma unroll
    for (int r = 0; r < 8; ++r) {
      const int m = row0 + r + ((lane >> 4) << 3);
      float v = acc[nt][r] + bv;
      if (RELU) v = v > 0.f ? v : 0.f;
      if (OUTBF) outB[(size_t)m * N + col] = (bf16)v;
      else       outF[(size_t)m * N + col] = v;
    }
  }
}

// ---------------------------------------------------------------------------
// Persistent LSTM scan: 16 WGs x 256 thr. WG g owns channels [g*32, g*32+32).
// Its 128 W_hh rows (q*512 + g*32 + local, q=0..3) are preloaded to LDS once
// via async copies and re-read every step. Wave layout: chgrp = wave&1
// (16-channel half), rb = wave>>1 (16-row block). Each wave accumulates the
// SAME channels for all four gates -> i,f,g,o co-resident per lane.
__global__ void __launch_bounds__(256)
lstm_scan(const float* __restrict__ gatesx, const bf16* __restrict__ Whh,
          const float* __restrict__ bhh, const float* __restrict__ h0,
          const float* __restrict__ c0, const int* __restrict__ masks,
          bf16* __restrict__ hbuf, bf16* __restrict__ hiddens,
          float* __restrict__ dout, int* __restrict__ bar) {
  extern __shared__ bf16 sWhh[];               // [128][512] = 128KB dynamic LDS
  const int g = blockIdx.x;
  const int tid = threadIdx.x;
  const int wave = tid >> 5, lane = tid & 31;
  const int chgrp = wave & 1;
  const int rb = wave >> 1;
  const int cbase = g * 32 + chgrp * 16;
  const int ch = cbase + (lane & 15);
  const int bo = rb * 16 + ((lane >> 4) << 3);  // +r gives this lane's batch rows

  // preload W_hh slice: local row lr = q*32 + local  <-  global row q*512+g*32+local
  for (int idx = tid; idx < 8192; idx += 256) {  // 8192 x 16B = 128KB
    const int lr = idx >> 6, part = idx & 63;    // 64 x 16B per 512-elem row
    const int grow = (lr >> 5) * C_ + g * 32 + (lr & 31);
    async_copy16(&Whh[(size_t)grow * C_ + part * 8], &sWhh[lr * C_ + part * 8]);
  }

  float c_reg[8], h_reg[8];
#pragma unroll
  for (int r = 0; r < 8; ++r) {
    c_reg[r] = c0[(bo + r) * C_ + ch];
    h_reg[r] = h0[(bo + r) * C_ + ch];
    hbuf[(bo + r) * C_ + ch] = (bf16)h_reg[r];   // buffer 0 = h at t=0 input
  }
  const float bi  = bhh[ch];
  const float bff = bhh[C_ + ch];
  const float bgg = bhh[2 * C_ + ch];
  const float boo = bhh[3 * C_ + ch];

  async_wait_all();
  int epoch = 1;
  grid_sync(bar, epoch * NWG_);                  // also covers LDS preload

  for (int t = 0; t < T_; ++t) {
    const bf16* hin  = hbuf + (size_t)(t & 1) * (B_ * C_);
    bf16*       hout = hbuf + (size_t)((t + 1) & 1) * (B_ * C_);
    const int* mrow = masks + t * B_;
    const int amask = mrow[rb * 16 + (lane & 15)];  // this lane's A-frag row

    if (t + 1 < T_) {                             // warm next step's gx stream
      const size_t gx1 = (size_t)((t + 1) * B_ + bo) * G4_;
      __builtin_prefetch(&gatesx[gx1 + ch], 0, 1);
      __builtin_prefetch(&gatesx[gx1 + C_ + ch], 0, 1);
      __builtin_prefetch(&gatesx[gx1 + 2 * C_ + ch], 0, 1);
      __builtin_prefetch(&gatesx[gx1 + 3 * C_ + ch], 0, 1);
    }

    f32x8 acc[4];
#pragma unroll
    for (int q = 0; q < 4; ++q)
#pragma unroll
      for (int i = 0; i < 8; ++i) acc[q][i] = 0.f;

    for (int kc = 0; kc < C_; kc += 32) {
      bf16x16 a = load_frag(hin + (size_t)(rb * 16) * C_ + kc, C_);
      if (amask) {                                  // h *= (1 - m_t)
#pragma unroll
        for (int i = 0; i < 16; ++i) a[i] = (bf16)0.f;
      }
#pragma unroll
      for (int q = 0; q < 4; ++q) {
        const bf16x16 bfr =
            load_frag(&sWhh[(q * 32 + chgrp * 16) * C_ + kc], C_);
        acc[q] = wmma_bf16(a, bfr, acc[q]);
      }
    }

#pragma unroll
    for (int r = 0; r < 8; ++r) {
      const int b = bo + r;
      const float keep = mrow[b] ? 0.f : 1.f;
      const size_t gx = (size_t)(t * B_ + b) * G4_;
      const float iv = fsigmoid(acc[0][r] + gatesx[gx + ch]          + bi);
      const float fv = fsigmoid(acc[1][r] + gatesx[gx + C_ + ch]     + bff);
      const float gv = ftanh   (acc[2][r] + gatesx[gx + 2 * C_ + ch] + bgg);
      const float ov = fsigmoid(acc[3][r] + gatesx[gx + 3 * C_ + ch] + boo);
      const float cc = fv * (c_reg[r] * keep) + iv * gv;
      const float hv = ov * ftanh(cc);
      c_reg[r] = cc;
      h_reg[r] = hv;
      hout[b * C_ + ch] = (bf16)hv;
      hiddens[(size_t)(t * B_ + b) * C_ + ch] = (bf16)hv;
    }
    grid_sync(bar, (++epoch) * NWG_);
  }

  // hT, cT -> d_out tail
#pragma unroll
  for (int r = 0; r < 8; ++r) {
    dout[2 * TBA_ + (size_t)(bo + r) * C_ + ch]           = h_reg[r];
    dout[2 * TBA_ + B_ * C_ + (size_t)(bo + r) * C_ + ch] = c_reg[r];
  }
}

// ---------------------------------------------------------------------------
// means = tanh(hiddens @ W_mean^T + b_mean); N = 32 (2 tiles per wave).
__global__ void __launch_bounds__(256)
gemm_means(const bf16* __restrict__ Hid, const bf16* __restrict__ Wm,
           const float* __restrict__ bm, float* __restrict__ out) {
  const int wave = threadIdx.x >> 5, lane = threadIdx.x & 31;
  const int row0 = blockIdx.x * 128 + wave * 16;
  f32x8 acc[2];
#pragma unroll
  for (int nt = 0; nt < 2; ++nt)
#pragma unroll
    for (int i = 0; i < 8; ++i) acc[nt][i] = 0.f;

  for (int kc = 0; kc < C_; kc += 32) {
    const bf16x16 a = load_frag(Hid + (size_t)row0 * C_ + kc, C_);
#pragma unroll
    for (int nt = 0; nt < 2; ++nt) {
      const bf16x16 b = load_frag(Wm + (size_t)(nt * 16) * C_ + kc, C_);
      acc[nt] = wmma_bf16(a, b, acc[nt]);
    }
  }
#pragma unroll
  for (int nt = 0; nt < 2; ++nt) {
    const int col = nt * 16 + (lane & 15);
    const float bv = bm[col];
#pragma unroll
    for (int r = 0; r < 8; ++r) {
      const int m = row0 + r + ((lane >> 4) << 3);
      out[(size_t)m * A_ + col] = ftanh(acc[nt][r] + bv);
    }
  }
}

// ---------------------------------------------------------------------------
extern "C" void kernel_launch(void* const* d_in, const int* in_sizes, int n_in,
                              void* d_out, int out_size, void* d_ws, size_t ws_size,
                              hipStream_t stream) {
  (void)in_sizes; (void)n_in; (void)out_size; (void)ws_size;

  const float* xs   = (const float*)d_in[0];
  const float* h0   = (const float*)d_in[1];
  const float* c0   = (const float*)d_in[2];
  const int*   hm   = (const int*)  d_in[3];
  const float* Win  = (const float*)d_in[4];
  const float* bin  = (const float*)d_in[5];
  const float* Wih  = (const float*)d_in[6];
  const float* bih  = (const float*)d_in[7];
  const float* Whh  = (const float*)d_in[8];
  const float* bhh  = (const float*)d_in[9];
  const float* Wm   = (const float*)d_in[10];
  const float* bm   = (const float*)d_in[11];
  const float* ls   = (const float*)d_in[12];
  float* out = (float*)d_out;

  char* ws = (char*)d_ws;
  bf16*  xs_b   = (bf16*)(ws + OFF_XSB);
  bf16*  Win_b  = (bf16*)(ws + OFF_WINB);
  bf16*  Wih_b  = (bf16*)(ws + OFF_WIHB);
  bf16*  Whh_b  = (bf16*)(ws + OFF_WHHB);
  bf16*  Wm_b   = (bf16*)(ws + OFF_WMB);
  bf16*  xproj  = (bf16*)(ws + OFF_XP);
  float* gatesx = (float*)(ws + OFF_GX);
  bf16*  hidden = (bf16*)(ws + OFF_HID);
  bf16*  hbuf   = (bf16*)(ws + OFF_HB);
  int*   bar    = (int*)(ws + OFF_BAR);

  constexpr int SCAN_LDS = 128 * C_ * 2;   // 128KB dynamic LDS (of 320KB/WGP)
  static int lds_attr_done = 0;
  if (!lds_attr_done) {   // attribute set is idempotent & capture-safe
    (void)hipFuncSetAttribute(reinterpret_cast<const void*>(&lstm_scan),
                              hipFuncAttributeMaxDynamicSharedMemorySize, SCAN_LDS);
    lds_attr_done = 1;
  }

  init_bar<<<1, 64, 0, stream>>>(bar);

  cvt_f32_bf16<<<1024, 256, 0, stream>>>(xs,  xs_b,  (long)TB_ * OBS_);
  cvt_f32_bf16<<<256,  256, 0, stream>>>(Win, Win_b, (long)H_ * OBS_);
  cvt_f32_bf16<<<1024, 256, 0, stream>>>(Wih, Wih_b, (long)G4_ * H_);
  cvt_f32_bf16<<<1024, 256, 0, stream>>>(Whh, Whh_b, (long)G4_ * C_);
  cvt_f32_bf16<<<64,   256, 0, stream>>>(Wm,  Wm_b,  (long)A_ * C_);

  // x_proj = relu(xs @ W_in^T + b_in)   : M=32768 K=128 N=1024
  gemm_bias_act<true, true><<<dim3(H_ / 128, TB_ / 128), 256, 0, stream>>>(
      xs_b, Win_b, bin, nullptr, xproj, TB_, H_, OBS_);

  // gates_x = x_proj @ W_ih^T + b_ih    : M=32768 K=1024 N=2048
  gemm_bias_act<false, false><<<dim3(G4_ / 128, TB_ / 128), 256, 0, stream>>>(
      xproj, Wih_b, bih, gatesx, nullptr, TB_, G4_, H_);

  // sequential scan (persistent, one grid barrier per step)
  lstm_scan<<<NWG_, 256, SCAN_LDS, stream>>>(gatesx, Whh_b, bhh, h0, c0, hm,
                                             hbuf, hidden, out, bar);

  // means = tanh(hiddens @ W_mean^T + b_mean)
  gemm_means<<<TB_ / 128, 256, 0, stream>>>(hidden, Wm_b, bm, out);

  // log_std broadcast
  fill_logstd<<<1024, 256, 0, stream>>>(ls, out + TBA_, TBA_);
}
